// STAGRegressor_46205258170707
// MI455X (gfx1250) — matrix operand order repaired and checked by reading, not compiled, not run on hardware
//
#include <hip/hip_runtime.h>
#include <hip/hip_bf16.h>
#include <math.h>

#define D_IN 590
#define H1N 128
#define H2N 64
#define KC1 19              // ceil(590/32) chunks of K for GEMM1
#define LDSA_STRIDE 34      // bf16 elements per row (32 + 2 pad) -> 17-bank row stride
#define LDSH_STRIDE 132     // bf16 elements per row (128 + 4 pad)

typedef __attribute__((ext_vector_type(16))) __bf16 v16bf;
typedef __attribute__((ext_vector_type(8)))  float  v8f;

union Frag { v16bf v; unsigned int u[8]; };

__device__ __forceinline__ unsigned short f2bf(float f) {
  union { float f; unsigned int u; } v; v.f = f;
  unsigned int u = v.u;
  return (unsigned short)((u + 0x7FFFu + ((u >> 16) & 1u)) >> 16); // RNE truncate
}

__device__ __forceinline__ float sigmoidf_(float x) { return 1.0f / (1.0f + expf(-x)); }

// Stage one 128x32 K-chunk of x and W1 into LDS as packed bf16 pairs.
// Full chunks (kc < 18): unguarded float2 loads + b32 LDS stores.
// Tail chunk (kc == 18): zero-fill K >= 590 (590 is even, no straddling pair).
__device__ __forceinline__ void stage_tiles(const float* __restrict__ x,
                                            const float* __restrict__ W1,
                                            unsigned short* __restrict__ sX,
                                            unsigned short* __restrict__ sW,
                                            int rowBlk, int kc, int t) {
  if (kc < 18) {
    #pragma unroll
    for (int i = 0; i < 8; ++i) {
      int idx = i * 256 + t;         // 2048 float2 slots = 128 rows x 16 pairs
      int r   = idx >> 4;
      int c2  = idx & 15;
      long xo = (long)(rowBlk + r) * D_IN + kc * 32 + c2 * 2;
      long wo = (long)r * D_IN + kc * 32 + c2 * 2;
      float2 xv = *(const float2*)(x + xo);
      float2 wv = *(const float2*)(W1 + wo);
      if (i == 0 && kc < 17) __builtin_prefetch(x + xo + 32, 0, 1); // +1 chunk ahead of staged
      unsigned int xp = (unsigned int)f2bf(xv.x) | ((unsigned int)f2bf(xv.y) << 16);
      unsigned int wp = (unsigned int)f2bf(wv.x) | ((unsigned int)f2bf(wv.y) << 16);
      *(unsigned int*)(sX + r * LDSA_STRIDE + c2 * 2) = xp;
      *(unsigned int*)(sW + r * LDSA_STRIDE + c2 * 2) = wp;
    }
  } else {
    #pragma unroll
    for (int i = 0; i < 8; ++i) {
      int idx = i * 256 + t;
      int r   = idx >> 4;
      int c2  = idx & 15;
      int kg  = 18 * 32 + c2 * 2;
      float2 xv; xv.x = 0.0f; xv.y = 0.0f;
      float2 wv; wv.x = 0.0f; wv.y = 0.0f;
      if (kg + 1 < D_IN) {
        xv = *(const float2*)(x + (long)(rowBlk + r) * D_IN + kg);
        wv = *(const float2*)(W1 + (long)r * D_IN + kg);
      }
      unsigned int xp = (unsigned int)f2bf(xv.x) | ((unsigned int)f2bf(xv.y) << 16);
      unsigned int wp = (unsigned int)f2bf(wv.x) | ((unsigned int)f2bf(wv.y) << 16);
      *(unsigned int*)(sX + r * LDSA_STRIDE + c2 * 2) = xp;
      *(unsigned int*)(sW + r * LDSA_STRIDE + c2 * 2) = wp;
    }
  }
}

__global__ __launch_bounds__(256)
void stag_fused_kernel(const float* __restrict__ x,   const float* __restrict__ u,
                       const float* __restrict__ W1,  const float* __restrict__ b1,
                       const float* __restrict__ W2,  const float* __restrict__ b2,
                       const float* __restrict__ Wmu, const float* __restrict__ bmu,
                       const float* __restrict__ Wsl, const float* __restrict__ bsl,
                       const float* __restrict__ Wsr, const float* __restrict__ bsr,
                       const float* __restrict__ a_raw, const float* __restrict__ b_raw,
                       float* __restrict__ out, int Btot)
{
  // Double-buffered x/W1 tiles + persistent h1 + dot scratch.
  // [sX0 8704][sX1 8704][sW0 8704][sW1 8704][sH 33792][sDot 2048] = 70656 B
  // sW2 (64x132 bf16 = 16896 B) aliases the sX region after GEMM1.
  __shared__ __align__(16) unsigned char smem[70656];
  unsigned short* sX0 = (unsigned short*)(smem);
  unsigned short* sX1 = (unsigned short*)(smem + 8704);
  unsigned short* sW0 = (unsigned short*)(smem + 17408);
  unsigned short* sW1t= (unsigned short*)(smem + 26112);
  unsigned short* sH  = (unsigned short*)(smem + 34816);
  unsigned short* sW2 = (unsigned short*)(smem);
  float*          sDot= (float*)(smem + 68608);

  const int t    = (int)threadIdx.x;
  const int lane = t & 31;
  const int wave = t >> 5;
  const int ln   = lane & 15;   // N (B/C/D) or M (A) index within tile
  const int lh   = lane >> 4;   // lane half
  const int rowBlk  = (int)blockIdx.x * 128;
  const int waveRow = wave * 16;

  // ---------------- GEMM1: h1 = relu(x @ W1^T + b1) ----------------
  v8f acc[8];
  #pragma unroll
  for (int i = 0; i < 8; ++i)
    #pragma unroll
    for (int j = 0; j < 8; ++j) acc[i][j] = 0.0f;

  stage_tiles(x, W1, sX0, sW0, rowBlk, 0, t);
  __syncthreads();

  for (int kc = 0; kc < KC1; ++kc) {
    const int p = kc & 1;
    unsigned short* cX = p ? sX1 : sX0;
    unsigned short* cW = p ? sW1t : sW0;
    // Prefetch-stage next chunk into the other buffer while this one computes.
    if (kc + 1 < KC1)
      stage_tiles(x, W1, p ? sX0 : sX1, p ? sW0 : sW1t, rowBlk, kc + 1, t);

    // A fragment: 16x32 bf16, lane m = ln; VGPR r -> K = 16*lh + 2r
    Frag a;
    #pragma unroll
    for (int r = 0; r < 8; ++r) {
      int k0 = (lh << 4) + (r << 1);
      a.u[r] = *(const unsigned int*)&cX[(waveRow + ln) * LDSA_STRIDE + k0];
    }
    #pragma unroll
    for (int nt = 0; nt < 8; ++nt) {
      // B fragment: 32x16 bf16 (KxN), lane n = ln; VGPR j -> K = 16*lh + 2j
      Frag b;
      #pragma unroll
      for (int j = 0; j < 8; ++j) {
        int k0 = (lh << 4) + (j << 1);
        b.u[j] = *(const unsigned int*)&cW[(nt * 16 + ln) * LDSA_STRIDE + k0];
      }
      acc[nt] = __builtin_amdgcn_wmma_f32_16x16x32_bf16(
          false, a.v, false, b.v, (short)0, acc[nt], false, false);
    }
    __syncthreads();   // next buffer staged AND this buffer free for overwrite
  }

  // bias + relu, spill h1 (bf16) to LDS in C/D layout: VGPR r -> row = r + 8*lh
  #pragma unroll
  for (int nt = 0; nt < 8; ++nt) {
    float bv = b1[nt * 16 + ln];
    #pragma unroll
    for (int r = 0; r < 8; ++r) {
      float h = acc[nt][r] + bv;
      h = h > 0.0f ? h : 0.0f;
      int row = waveRow + (lh << 3) + r;
      sH[row * LDSH_STRIDE + nt * 16 + ln] = f2bf(h);
    }
  }
  __syncthreads();

  // Stage all of W2 (64x128) as bf16 into the aliased region.
  #pragma unroll
  for (int i = 0; i < 32; ++i) {
    int idx = i * 256 + t;
    int n = idx >> 7;
    int k = idx & 127;
    sW2[n * LDSH_STRIDE + k] = f2bf(W2[n * 128 + k]);
  }
  __syncthreads();

  // ---------------- GEMM2: h2 = relu(h1 @ W2^T + b2) ----------------
  v8f acc2[4];
  #pragma unroll
  for (int i = 0; i < 4; ++i)
    #pragma unroll
    for (int j = 0; j < 8; ++j) acc2[i][j] = 0.0f;

  #pragma unroll
  for (int kc = 0; kc < 4; ++kc) {
    Frag a;
    #pragma unroll
    for (int r = 0; r < 8; ++r) {
      int k0 = kc * 32 + (lh << 4) + (r << 1);
      a.u[r] = *(const unsigned int*)&sH[(waveRow + ln) * LDSH_STRIDE + k0];
    }
    #pragma unroll
    for (int nt = 0; nt < 4; ++nt) {
      Frag b;
      #pragma unroll
      for (int j = 0; j < 8; ++j) {
        int k0 = kc * 32 + (lh << 4) + (j << 1);
        b.u[j] = *(const unsigned int*)&sW2[(nt * 16 + ln) * LDSH_STRIDE + k0];
      }
      acc2[nt] = __builtin_amdgcn_wmma_f32_16x16x32_bf16(
          false, a.v, false, b.v, (short)0, acc2[nt], false, false);
    }
  }

  // ---------------- head dots: mu / log_sl / log_sr ----------------
  float dm[8], dsl_[8], dsr_[8];
  #pragma unroll
  for (int r = 0; r < 8; ++r) { dm[r] = 0.0f; dsl_[r] = 0.0f; dsr_[r] = 0.0f; }

  #pragma unroll
  for (int nt = 0; nt < 4; ++nt) {
    int f = nt * 16 + ln;
    float bv = b2[f], wm = Wmu[f], wl = Wsl[f], wr = Wsr[f];
    #pragma unroll
    for (int r = 0; r < 8; ++r) {
      float h = acc2[nt][r] + bv;
      h = h > 0.0f ? h : 0.0f;
      dm[r]   += h * wm;
      dsl_[r] += h * wl;
      dsr_[r] += h * wr;
    }
  }
  // xor-reduce over the 16-lane half (rows m split by half in C-layout)
  #pragma unroll
  for (int off = 1; off < 16; off <<= 1) {
    #pragma unroll
    for (int r = 0; r < 8; ++r) {
      dm[r]   += __shfl_xor(dm[r],   off, 32);
      dsl_[r] += __shfl_xor(dsl_[r], off, 32);
      dsr_[r] += __shfl_xor(dsr_[r], off, 32);
    }
  }
  if (ln == 0) {
    #pragma unroll
    for (int r = 0; r < 8; ++r) {
      int row = waveRow + (lh << 3) + r;
      sDot[row * 4 + 0] = dm[r];
      sDot[row * 4 + 1] = dsl_[r];
      sDot[row * 4 + 2] = dsr_[r];
    }
  }
  __syncthreads();

  // ---------------- per-row STAG epilogue ----------------
  if (t < 128) {
    const float SQRT2     = 1.41421356237309515f;
    const float INV_SQRT2 = 0.70710678118654752f;
    const float SQRT2PI   = 2.50662827463100024f;
    const float EPS       = 1e-6f;
    int row = rowBlk + t;

    float mu  = sDot[t * 4 + 0] + bmu[0];
    float lsl = sDot[t * 4 + 1] + bsl[0];
    float lsr = sDot[t * 4 + 2] + bsr[0];
    lsl = fminf(fmaxf(lsl, -6.0f), 2.0f);
    lsr = fminf(fmaxf(lsr, -6.0f), 2.0f);
    float sl = expf(lsl) + EPS;
    float sr = expf(lsr) + EPS;

    float a = -0.1f + sigmoidf_(a_raw[0]) * 0.2f;
    float b =  0.9f + sigmoidf_(b_raw[0]) * 0.2f;
    mu = fminf(fmaxf(mu, a), b);

    float Phi_ma = 0.5f * (1.0f + erff((mu - a) / sl * INV_SQRT2));
    float Phi_bm = 0.5f * (1.0f + erff((b - mu) / sr * INV_SQRT2));
    float tL = sl * (Phi_ma - 0.5f);
    float tR = sr * (Phi_bm - 0.5f);
    float Z  = SQRT2PI * (tL + tR);
    float pL = SQRT2PI * tL / Z;

    float uc = fminf(fmaxf(u[row], EPS), 1.0f - EPS);
    float ul = fminf(fmaxf(uc / pL, EPS), 1.0f - EPS);
    float tl = fminf(fmaxf(Phi_ma - ul * (Phi_ma - 0.5f), EPS), 1.0f - EPS);
    float zl = mu - sl * (erfinvf(2.0f * tl - 1.0f) * SQRT2);
    float ur = fminf(fmaxf((uc - pL) / (1.0f - pL), EPS), 1.0f - EPS);
    float tr = fminf(fmaxf(0.5f + ur * (Phi_bm - 0.5f), EPS), 1.0f - EPS);
    float zr = mu + sr * (erfinvf(2.0f * tr - 1.0f) * SQRT2);
    float z  = (uc <= pL) ? zl : zr;
    float y  = fminf(fmaxf(z, a), b);

    out[row]             = y;
    out[Btot + row]      = mu;
    out[2 * Btot + row]  = sl;
    out[3 * Btot + row]  = sr;
  }
}

extern "C" void kernel_launch(void* const* d_in, const int* in_sizes, int n_in,
                              void* d_out, int out_size, void* d_ws, size_t ws_size,
                              hipStream_t stream) {
  const float* x     = (const float*)d_in[0];
  const float* u     = (const float*)d_in[1];
  const float* W1    = (const float*)d_in[2];
  const float* b1    = (const float*)d_in[3];
  const float* W2    = (const float*)d_in[4];
  const float* b2    = (const float*)d_in[5];
  const float* Wmu   = (const float*)d_in[6];
  const float* bmu   = (const float*)d_in[7];
  const float* Wsl   = (const float*)d_in[8];
  const float* bsl   = (const float*)d_in[9];
  const float* Wsr   = (const float*)d_in[10];
  const float* bsr   = (const float*)d_in[11];
  const float* a_raw = (const float*)d_in[12];
  const float* b_raw = (const float*)d_in[13];

  int Btot = in_sizes[0] / D_IN;              // 131072
  dim3 grid((unsigned)(Btot / 128));          // 128 rows per block
  dim3 block(256);                            // 8 waves (wave32)
  stag_fused_kernel<<<grid, block, 0, stream>>>(
      x, u, W1, b1, W2, b2, Wmu, bmu, Wsl, bsl, Wsr, bsr, a_raw, b_raw,
      (float*)d_out, Btot);
}
